// SAGE_37675453120774
// MI455X (gfx1250) — compile-verified, matrix operand order checked by reading, not run on hardware
//
#include <hip/hip_runtime.h>
#include <hip/hip_bf16.h>

// ---------------------------------------------------------------------------
// GraphSAGE (pool + mean) forward for MI455X / gfx1250.
// GEMMs via v_wmma_f32_16x16x32_bf16 with weights pre-packed into the WMMA
// B-fragment lane layout (2 x b128 loads per fragment); A tiles staged to LDS
// via GLOBAL_LOAD_ASYNC_TO_LDS_B128 (ASYNCcnt-tracked direct-to-LDS path).
// Edge aggregation via L2 atomics with bf16-compressed gathers.
// ---------------------------------------------------------------------------

typedef __attribute__((ext_vector_type(16))) __bf16 v16bf;
typedef __attribute__((ext_vector_type(8)))  float  v8f;

#define IN_F   128
#define TILE_M 16

__device__ __forceinline__ unsigned short f2bf(float f) {
  unsigned u = __float_as_uint(f);
  u += 0x7FFFu + ((u >> 16) & 1u);        // round-to-nearest-even
  return (unsigned short)(u >> 16);
}
__device__ __forceinline__ float bf2f(unsigned short s) {
  return __uint_as_float(((unsigned)s) << 16);
}

union Frag16 { v16bf v; unsigned int u[8]; uint4 q[2]; };

// ----------------------- async LDS staging (CDNA5) -------------------------
// Builtin params are v4i pointers in addrspace(1)/(3) (from hipcc diagnostic).

#if __has_builtin(__builtin_amdgcn_global_load_async_to_lds_b128)
#define HAVE_ASYNC_LDS 1
typedef int v4i_vec __attribute__((vector_size(16)));
typedef __attribute__((address_space(1))) v4i_vec* g_v4i_ptr;
typedef __attribute__((address_space(3))) v4i_vec* l_v4i_ptr;
__device__ __forceinline__ void async_ld_b128(void* lds, const void* g) {
  __builtin_amdgcn_global_load_async_to_lds_b128(
      (g_v4i_ptr)(g), (l_v4i_ptr)(lds), 0, 0);
}
#endif

__device__ __forceinline__ void wait_async0() {
#if __has_builtin(__builtin_amdgcn_s_wait_asynccnt)
  __builtin_amdgcn_s_wait_asynccnt(0);
#else
  asm volatile("s_wait_asynccnt 0x0" ::: "memory");
#endif
}

// ------------------------------- utility kernels ---------------------------

__global__ void zero_kernel(float* a, float* b, long na, long nb) {
  long i = (long)blockIdx.x * blockDim.x + threadIdx.x;
  if (i < na) a[i] = 0.0f;
  if (i < nb) b[i] = 0.0f;
}

__global__ void cast_kernel(const float* __restrict__ in,
                            unsigned short* __restrict__ out, long n) {
  long i = (long)blockIdx.x * blockDim.x + threadIdx.x;
  if (i < n) out[i] = f2bf(in[i]);
}

// cast accumulator -> bf16, and re-zero accumulator for the next segment op
__global__ void castzero_kernel(float* __restrict__ acc,
                                unsigned short* __restrict__ out, long n) {
  long i = (long)blockIdx.x * blockDim.x + threadIdx.x;
  if (i < n) { out[i] = f2bf(acc[i]); acc[i] = 0.0f; }
}

__global__ void mean_kernel(const float* __restrict__ acc,
                            const float* __restrict__ deg,
                            unsigned short* __restrict__ out, long n) {
  long i = (long)blockIdx.x * blockDim.x + threadIdx.x;
  if (i < n) {
    float d = deg[i >> 7];                       // 128 feats per row
    out[i] = f2bf(acc[i] / fmaxf(d, 1.0f));
  }
}

__global__ void deg_kernel(const int* __restrict__ dst, float* deg, int E) {
  int e = blockIdx.x * blockDim.x + threadIdx.x;
  if (e < E) atomicAdd(&deg[dst[e]], 1.0f);
}

// Pack a 128x128 fp32 weight matrix into bf16 WMMA B-fragment layout.
// Index: (((ks*8 + strip)*32 + lane)*8 + dword)  -> one u32 = bf16 pair.
// For lane l (g=l>>4, n=l&15), dword v (e=2v):
//   K = 32*ks + ((e&7) | ((e>>3)<<4) | (g<<3));  pair = W[K][16*strip+n], W[K+1][...]
__global__ void pack_w_kernel(const float* __restrict__ W,
                              unsigned int* __restrict__ out) {
  int i = blockIdx.x * blockDim.x + threadIdx.x;
  if (i >= 8192) return;
  int v  = i & 7;
  int l  = (i >> 3) & 31;
  int w  = (i >> 8) & 7;
  int ks = i >> 11;
  int g = l >> 4, n = l & 15;
  int e = 2 * v;
  int K = ks * 32 + ((e & 7) | ((e >> 3) << 4) | (g << 3));
  unsigned lo = f2bf(W[K * IN_F + 16 * w + n]);
  unsigned hi = f2bf(W[(K + 1) * IN_F + 16 * w + n]);
  out[i] = lo | (hi << 16);
}

// ------------------------------ edge scatter kernels -----------------------
// One wave32 per edge; 4 features per lane (8-byte bf16 gather per lane).

__global__ __launch_bounds__(256)
void smax_kernel(const unsigned short* __restrict__ hp,
                 const int* __restrict__ src, const int* __restrict__ dst,
                 unsigned int* __restrict__ mx, int E) {
  int e = (blockIdx.x * 256 + threadIdx.x) >> 5;
  int lane = threadIdx.x & 31;
  if (e >= E) return;
  int s = src[e], d = dst[e];
  uint2 p = ((const uint2*)(hp + (size_t)s * IN_F))[lane];
  unsigned int* o = mx + (size_t)d * IN_F + lane * 4;
  // hp >= 0 after relu => fp32 bit pattern is monotone as uint
  atomicMax(o + 0, __float_as_uint(bf2f((unsigned short)(p.x & 0xffffu))));
  atomicMax(o + 1, __float_as_uint(bf2f((unsigned short)(p.x >> 16))));
  atomicMax(o + 2, __float_as_uint(bf2f((unsigned short)(p.y & 0xffffu))));
  atomicMax(o + 3, __float_as_uint(bf2f((unsigned short)(p.y >> 16))));
}

__global__ __launch_bounds__(256)
void ssum_kernel(const unsigned short* __restrict__ h,
                 const int* __restrict__ src, const int* __restrict__ dst,
                 float* __restrict__ sm, int E) {
  int e = (blockIdx.x * 256 + threadIdx.x) >> 5;
  int lane = threadIdx.x & 31;
  if (e >= E) return;
  int s = src[e], d = dst[e];
  uint2 p = ((const uint2*)(h + (size_t)s * IN_F))[lane];
  float* o = sm + (size_t)d * IN_F + lane * 4;
  atomicAdd(o + 0, bf2f((unsigned short)(p.x & 0xffffu)));
  atomicAdd(o + 1, bf2f((unsigned short)(p.x >> 16)));
  atomicAdd(o + 2, bf2f((unsigned short)(p.y & 0xffffu)));
  atomicAdd(o + 3, bf2f((unsigned short)(p.y >> 16)));
}

// ------------------------------- WMMA GEMM ---------------------------------
// D[M x 128] = A1*W1 (+ A2*W2) + bias, with fused epilogue.
// W1/W2 are pre-packed B-fragment arrays (see pack_w_kernel).
// MODE 0: bias + relu, bf16 out (fc_pool)
// MODE 1: bias + row-L2-normalize + relu, bf16 out (conv1)
// MODE 2: bias + relu, fp32 out (conv2 -> d_out)
// One block = 16 rows x 128 cols; 8 waves, each owns a 16-col strip.

template <int MODE>
__global__ __launch_bounds__(256)
void gemm_kernel(const unsigned short* __restrict__ A1,
                 const unsigned short* __restrict__ A2,
                 const unsigned int* __restrict__ W1,
                 const unsigned int* __restrict__ W2,
                 const float* __restrict__ bias,
                 void* __restrict__ out, int M) {
  __shared__ unsigned short As1[TILE_M * IN_F];
  __shared__ unsigned short As2[TILE_M * IN_F];
  __shared__ float Dt[TILE_M * IN_F];
  __shared__ float rinv[TILE_M];

  const int t = threadIdx.x;
  const int rowbase = blockIdx.x * TILE_M;
  const bool full = (rowbase + TILE_M) <= M;   // block-uniform fast path

  // Stage A tiles in LDS: 16 rows x 256B; thread t -> row t/16, 16B chunk t%16.
  {
    int rl = t >> 4, c4 = t & 15;
    int row = rowbase + rl;
    unsigned short* l1 = As1 + rl * IN_F + c4 * 8;
    unsigned short* l2 = As2 + rl * IN_F + c4 * 8;
#ifdef HAVE_ASYNC_LDS
    if (row < M) {
      async_ld_b128(l1, A1 + (size_t)row * IN_F + c4 * 8);
      if (MODE != 0) async_ld_b128(l2, A2 + (size_t)row * IN_F + c4 * 8);
    } else {
      uint4 z = {0, 0, 0, 0};
      *(uint4*)l1 = z;
      if (MODE != 0) *(uint4*)l2 = z;
    }
    wait_async0();
#else
    uint4 za = {0, 0, 0, 0}, zb = {0, 0, 0, 0};
    if (row < M) {
      za = ((const uint4*)(A1 + (size_t)row * IN_F))[c4];
      if (MODE != 0) zb = ((const uint4*)(A2 + (size_t)row * IN_F))[c4];
    }
    *(uint4*)l1 = za;
    if (MODE != 0) *(uint4*)l2 = zb;
#endif
  }
  __syncthreads();

  const int wave = t >> 5, lane = t & 31;
  const int g = lane >> 4;        // K-half select (ISA 16-bit A layout)
  const int nh = lane & 15;       // row (A) / col (B,C,D) within tile
  const int arow = nh * IN_F;

  v8f acc = {};
#pragma unroll
  for (int ks = 0; ks < IN_F / 32; ++ks) {
    const int k0 = ks * 32;
    Frag16 a, b;
    // A fragment: two contiguous b128 loads from LDS (K-halves per lane group)
    a.q[0] = *(const uint4*)&As1[arow + k0 + 8 * g];
    a.q[1] = *(const uint4*)&As1[arow + k0 + 16 + 8 * g];
    // B fragment: two b128 loads from the pre-packed weight array
    const uint4* qb = (const uint4*)(W1 + (((ks * 8 + wave) * 32 + lane) << 3));
    b.q[0] = qb[0];
    b.q[1] = qb[1];
    acc = __builtin_amdgcn_wmma_f32_16x16x32_bf16(false, a.v, false, b.v,
                                                  (short)0, acc, false, false);
    if (MODE != 0) {
      Frag16 a2, b2;
      a2.q[0] = *(const uint4*)&As2[arow + k0 + 8 * g];
      a2.q[1] = *(const uint4*)&As2[arow + k0 + 16 + 8 * g];
      const uint4* qb2 = (const uint4*)(W2 + (((ks * 8 + wave) * 32 + lane) << 3));
      b2.q[0] = qb2[0];
      b2.q[1] = qb2[1];
      acc = __builtin_amdgcn_wmma_f32_16x16x32_bf16(false, a2.v, false, b2.v,
                                                    (short)0, acc, false, false);
    }
  }

  const int col = wave * TILE_M + nh;
  const float bv = bias[col];

  if (MODE == 1) {
    // Row L2 normalization needs whole rows: round-trip through LDS.
#pragma unroll
    for (int r = 0; r < 8; ++r) {
      int m = r + 8 * g;                  // C/D layout: m = vgpr + 8*(lane>=16)
      Dt[m * IN_F + col] = acc[r] + bv;
    }
    __syncthreads();
    if (t < TILE_M) {
      float s = 0.0f;
      for (int c = 0; c < IN_F; ++c) { float x = Dt[t * IN_F + c]; s += x * x; }
      rinv[t] = 1.0f / fmaxf(sqrtf(s), 1e-12f);
    }
    __syncthreads();
    // Each thread owns 8 consecutive feats of one row: pack into one b128 store.
    int idx0 = t * 8;                     // 256 threads * 8 = 16*128
    int m = idx0 >> 7;
    if (full || (rowbase + m) < M) {
      float sc = rinv[m];
      unsigned int pk[4];
#pragma unroll
      for (int j = 0; j < 4; ++j) {
        float v0 = Dt[idx0 + 2 * j] * sc;
        float v1 = Dt[idx0 + 2 * j + 1] * sc;
        v0 = v0 > 0.0f ? v0 : 0.0f;
        v1 = v1 > 0.0f ? v1 : 0.0f;
        pk[j] = (unsigned)f2bf(v0) | ((unsigned)f2bf(v1) << 16);
      }
      uint4 o4 = {pk[0], pk[1], pk[2], pk[3]};
      *(uint4*)((unsigned short*)out + (size_t)rowbase * IN_F + idx0) = o4;
    }
  } else {
    if (full) {
      // fast path: no per-row guards -> single base + immediate offsets
#pragma unroll
      for (int r = 0; r < 8; ++r) {
        int row = rowbase + r + 8 * g;
        float v = acc[r] + bv;
        v = v > 0.0f ? v : 0.0f;
        if (MODE == 0)
          ((unsigned short*)out)[(size_t)row * IN_F + col] = f2bf(v);
        else
          ((float*)out)[(size_t)row * IN_F + col] = v;
      }
    } else {
#pragma unroll
      for (int r = 0; r < 8; ++r) {
        int row = rowbase + r + 8 * g;
        if (row < M) {
          float v = acc[r] + bv;
          v = v > 0.0f ? v : 0.0f;
          if (MODE == 0)
            ((unsigned short*)out)[(size_t)row * IN_F + col] = f2bf(v);
          else
            ((float*)out)[(size_t)row * IN_F + col] = v;
        }
      }
    }
  }
}

// ------------------------------- launcher ----------------------------------

extern "C" void kernel_launch(void* const* d_in, const int* in_sizes, int n_in,
                              void* d_out, int out_size, void* d_ws, size_t ws_size,
                              hipStream_t stream) {
  const float* x      = (const float*)d_in[0];
  const int*   src    = (const int*)d_in[1];
  const int*   dst    = (const int*)d_in[2];
  const float* W_pool = (const float*)d_in[3];
  const float* b_pool = (const float*)d_in[4];
  const float* W_s1   = (const float*)d_in[5];
  const float* W_n1   = (const float*)d_in[6];
  const float* b1     = (const float*)d_in[7];
  const float* W_s2   = (const float*)d_in[8];
  const float* W_n2   = (const float*)d_in[9];
  const float* b2     = (const float*)d_in[10];

  const int N  = in_sizes[0] / IN_F;
  const int E  = in_sizes[1];
  const long nf = (long)N * IN_F;
  const int WPK = 8192;                       // u32s per packed 128x128 matrix

  // workspace layout (bf16 activations + one fp32 accumulator, reused)
  char* p = (char*)d_ws;
  unsigned short* x_bf    = (unsigned short*)p; p += nf * 2;
  unsigned short* hp_bf   = (unsigned short*)p; p += nf * 2;
  unsigned short* mx_bf   = (unsigned short*)p; p += nf * 2;
  unsigned short* h_bf    = (unsigned short*)p; p += nf * 2;
  unsigned short* mean_bf = (unsigned short*)p; p += nf * 2;
  float*          acc     = (float*)p;          p += nf * 4;   // max, then sum
  float*          deg     = (float*)p;          p += (size_t)N * 4;
  unsigned int*   wpk     = (unsigned int*)p;                  // 5 * 32KB

  unsigned int* wp  = wpk + 0 * WPK;
  unsigned int* ws1 = wpk + 1 * WPK;
  unsigned int* wn1 = wpk + 2 * WPK;
  unsigned int* ws2 = wpk + 3 * WPK;
  unsigned int* wn2 = wpk + 4 * WPK;

  const int T = 256;
  const unsigned gz = (unsigned)((nf + T - 1) / T);
  const unsigned gw = (unsigned)((WPK + T - 1) / T);
  const unsigned ge = (unsigned)(((long)E * 32 + T - 1) / T);   // 1 wave/edge
  const unsigned mb = (unsigned)((N + TILE_M - 1) / TILE_M);

  // init accumulators
  zero_kernel<<<gz, T, 0, stream>>>(acc, deg, nf, (long)N);

  // bf16 cast of x; weights cast+packed into WMMA B-fragment layout
  cast_kernel<<<gz, T, 0, stream>>>(x, x_bf, nf);
  pack_w_kernel<<<gw, T, 0, stream>>>(W_pool, wp);
  pack_w_kernel<<<gw, T, 0, stream>>>(W_s1,   ws1);
  pack_w_kernel<<<gw, T, 0, stream>>>(W_n1,   wn1);
  pack_w_kernel<<<gw, T, 0, stream>>>(W_s2,   ws2);
  pack_w_kernel<<<gw, T, 0, stream>>>(W_n2,   wn2);

  // in-degree
  deg_kernel<<<(E + T - 1) / T, T, 0, stream>>>(dst, deg, E);

  // hp = relu(x @ W_pool + b_pool)   (bf16 out for cheap gathers)
  gemm_kernel<0><<<mb, T, 0, stream>>>(x_bf, nullptr, wp, nullptr, b_pool,
                                       hp_bf, N);

  // mx = segment_max(hp[src], dst)  (uint atomic-max on nonneg floats)
  smax_kernel<<<ge, T, 0, stream>>>(hp_bf, src, dst, (unsigned int*)acc, E);
  castzero_kernel<<<gz, T, 0, stream>>>(acc, mx_bf, nf);

  // h = relu(normalize(x @ W_self1 + mx @ W_neigh1 + b1))
  gemm_kernel<1><<<mb, T, 0, stream>>>(x_bf, mx_bf, ws1, wn1, b1, h_bf, N);

  // mean = segment_sum(h[src], dst) / max(deg, 1)
  ssum_kernel<<<ge, T, 0, stream>>>(h_bf, src, dst, acc, E);
  mean_kernel<<<gz, T, 0, stream>>>(acc, deg, mean_bf, nf);

  // out = relu(h @ W_self2 + mean @ W_neigh2 + b2)   (fp32)
  gemm_kernel<2><<<mb, T, 0, stream>>>(h_bf, mean_bf, ws2, wn2, b2, d_out, N);
}
